// MockRecommenderSystem_2052994368000
// MI455X (gfx1250) — matrix operand order repaired and checked by reading, not compiled, not run on hardware
//
#include <hip/hip_runtime.h>
#include <hip/hip_bf16.h>

#define NUM_ITEMS 100000
#define TOP_K     50
#define EMBED_DIM 128
#define BATCH     1024
#define N_CHUNKS  16                 // item-dimension split for gemm2
#define N_TILES   (NUM_ITEMS / 16)   // 6250

typedef __attribute__((ext_vector_type(16))) __bf16 v16bf;
typedef __attribute__((ext_vector_type(8)))  __bf16 v8bf;
typedef __attribute__((ext_vector_type(8)))  float  v8f;
typedef __attribute__((ext_vector_type(4)))  float  f32x4;

// ---------------- ws layout (bytes) ----------------
#define WS_EBF_OFF    0            // Ebf  : 100000*128 bf16 (row-major, GEMM2 B)
#define WS_ETBF_OFF   25600000     // ETbf : 128*100000 bf16 (transposed, GEMM1 B)
#define WS_XEMB_OFF   51200000     // xemb : 1024*128 f32 (split-K accumulator)
#define WS_XEMBBF_OFF 51724288     // xembbf: 1024*128 bf16 (GEMM2 A)
#define WS_CANDS_OFF  51986432     // candS: 1024*16*50 f32
#define WS_CANDI_OFF  55263232     // candI: 1024*16*50 i32  (total ~58.5 MB)

// ---------------------------------------------------------------------------
__global__ void rec_zero_xemb(float* __restrict__ xemb) {
    int i = blockIdx.x * blockDim.x + threadIdx.x;
    if (i < BATCH * EMBED_DIM) xemb[i] = 0.0f;
}

// ---------------------------------------------------------------------------
// Convert E (f32, 100000x128) to bf16 twice: row-major Ebf and transposed ETbf.
// Both stay L2-resident (25.6 MB each vs 192 MB L2).
// ---------------------------------------------------------------------------
__global__ void rec_convert_E(const float* __restrict__ E,
                              __bf16* __restrict__ Ebf,
                              __bf16* __restrict__ ETbf) {
    size_t stride = (size_t)gridDim.x * blockDim.x;
    size_t total  = (size_t)NUM_ITEMS * EMBED_DIM;
    for (size_t i = (size_t)blockIdx.x * blockDim.x + threadIdx.x; i < total; i += stride) {
        size_t item = i >> 7;          // / 128
        size_t n    = i & 127;         // % 128
        __bf16 b = (__bf16)E[i];
        Ebf[i] = b;
        ETbf[n * (size_t)NUM_ITEMS + item] = b;
    }
}

// ---------------------------------------------------------------------------
// GEMM1: xemb(1024x128) += x @ E, bf16 WMMA, split-K, M=32 per wave.
// blockIdx.x = 32-row M-group (32), blockIdx.y = K-chunk (125, 25 k-steps).
// Every B fragment (from L2) is reused for two M-subtiles, halving L2 traffic
// vs 16-row blocking. x (409.6 MB) streams from HBM exactly once (~17.6 us
// floor at 23.3 TB/s) and is converted f32->bf16 in registers.
// ---------------------------------------------------------------------------
__global__ __launch_bounds__(32)
void rec_gemm1(const float* __restrict__ x,
               const __bf16* __restrict__ ETbf,
               float* __restrict__ xemb) {
    const int lane = threadIdx.x;
    const int m    = lane & 15;
    const int half = lane >> 4;
    const int m_base = blockIdx.x * 32;
    const int k0     = blockIdx.y * 800;     // 25 k-steps of 32

    v8f acc[2][8];
#pragma unroll
    for (int s = 0; s < 2; ++s)
#pragma unroll
        for (int t = 0; t < 8; ++t) acc[s][t] = (v8f){};

    const float* xrow0 = x + (size_t)(m_base + m) * NUM_ITEMS;
    const float* xrow1 = x + (size_t)(m_base + 16 + m) * NUM_ITEMS;

    for (int it = 0; it < 25; ++it) {
        const int k = k0 + it * 32;
        v16bf a[2];
#pragma unroll
        for (int s = 0; s < 2; ++s) {
            const float* xr = s ? xrow1 : xrow0;
            f32x4 a0 = *(const f32x4*)(xr + k + half * 16);
            f32x4 a1 = *(const f32x4*)(xr + k + half * 16 + 4);
            f32x4 a2 = *(const f32x4*)(xr + k + half * 16 + 8);
            f32x4 a3 = *(const f32x4*)(xr + k + half * 16 + 12);
#pragma unroll
            for (int e = 0; e < 4; ++e) {
                a[s][e]      = (__bf16)a0[e];
                a[s][4 + e]  = (__bf16)a1[e];
                a[s][8 + e]  = (__bf16)a2[e];
                a[s][12 + e] = (__bf16)a3[e];
            }
        }
#pragma unroll
        for (int t = 0; t < 8; ++t) {
            const int n = t * 16 + m;   // B lane column (embed index)
            v16bf b = *(const v16bf*)(ETbf + (size_t)n * NUM_ITEMS + k + half * 16);
            acc[0][t] = __builtin_amdgcn_wmma_f32_16x16x32_bf16(
                false, a[0], false, b, (short)0, acc[0][t], false, false);
            acc[1][t] = __builtin_amdgcn_wmma_f32_16x16x32_bf16(
                false, a[1], false, b, (short)0, acc[1][t], false, false);
        }
    }

    // C layout: vgpr r, lane -> row = r + 8*half, col = m (per N-tile).
#pragma unroll
    for (int s = 0; s < 2; ++s)
#pragma unroll
        for (int t = 0; t < 8; ++t)
#pragma unroll
            for (int r = 0; r < 8; ++r) {
                const int row = m_base + s * 16 + r + 8 * half;
                const int col = t * 16 + m;
                atomicAdd(&xemb[(size_t)row * EMBED_DIM + col], acc[s][t][r]);
            }
}

// ---------------------------------------------------------------------------
__global__ void rec_xemb_to_bf16(const float* __restrict__ xemb,
                                 __bf16* __restrict__ xembbf) {
    int i = blockIdx.x * blockDim.x + threadIdx.x;
    if (i < BATCH * EMBED_DIM) xembbf[i] = (__bf16)xemb[i];
}

// ---------------------------------------------------------------------------
// Wave-parallel argmax over (v, p); all 32 lanes end with the winning pair.
// ---------------------------------------------------------------------------
__device__ __forceinline__ void wave_argmax(float& v, int& p) {
#pragma unroll
    for (int off = 16; off >= 1; off >>= 1) {
        float ov = __shfl_xor(v, off, 32);
        int   op = __shfl_xor(p, off, 32);
        if (ov > v) { v = ov; p = op; }
    }
}

// ---------------------------------------------------------------------------
// GEMM2 + fused partial top-k, M=32 per block. Grid (32 M-groups x 16 chunks)
// x 256 thr = 512 blocks / 4096 waves (2 blocks per WGP at ~118 KB LDS).
// Each wave computes a 32x16 score tile (2 M-subtiles x 4 bf16 WMMAs, K=128),
// stages it in per-wave LDS, then ALL 32 lanes scan one row each (no EXEC
// divergence) maintaining per-row top-50 lists. Block merges its 8 wave lists
// per row via wave-parallel argmax into per-(row, chunk) candidates. B
// fragments are reused across both M-subtiles, halving L2 traffic. The 409 MB
// score tensor never touches HBM.
// Lists laid out [32 rows][8 waves][50] so a row's 400 candidates are
// contiguous for the merge.
// ---------------------------------------------------------------------------
__global__ __launch_bounds__(256)
void rec_gemm2_topk(const __bf16* __restrict__ xembbf,
                    const __bf16* __restrict__ Ebf,
                    float* __restrict__ candS,
                    int* __restrict__ candI) {
    extern __shared__ float smem[];
    float* staging = smem;                                        // 8 waves * 512 f32
    float* lscore  = smem + 8 * 512;                              // [32][8][50] f32
    int*   lidx    = (int*)(smem + 8 * 512 + 32 * 8 * TOP_K);     // [32][8][50] i32

    const int tid   = threadIdx.x;
    const int wave  = tid >> 5;
    const int lane  = tid & 31;
    const int m     = lane & 15;
    const int half  = lane >> 4;
    const int m_base = blockIdx.x * 32;
    const int chunk  = blockIdx.y;

    for (int i = tid; i < 32 * 8 * TOP_K; i += 256) {
        lscore[i] = -__builtin_inff();
        lidx[i]   = -1;
    }
    __syncthreads();

    // A fragments: 2 M-subtiles x 4 x (16x32 bf16) covering K = 0..127.
    v16bf afr[2][4];
#pragma unroll
    for (int s = 0; s < 2; ++s) {
        const __bf16* arow = xembbf + (size_t)(m_base + s * 16 + m) * EMBED_DIM;
#pragma unroll
        for (int f = 0; f < 4; ++f) {
            v8bf lo = *(const v8bf*)(arow + f * 32 + half * 16);
            v8bf hi = *(const v8bf*)(arow + f * 32 + half * 16 + 8);
            v16bf a;
#pragma unroll
            for (int e = 0; e < 8; ++e) { a[e] = lo[e]; a[8 + e] = hi[e]; }
            afr[s][f] = a;
        }
    }

    float* mystage = staging + wave * 512;          // 32 rows x 16 cols
    float* myls    = lscore + (lane * 8 + wave) * TOP_K;   // lane == row-in-block
    int*   myli    = lidx   + (lane * 8 + wave) * TOP_K;
    float  curMin  = -__builtin_inff();
    int    minPos  = 0;
    const int stream_id = chunk * 8 + wave;          // 0..127

    for (int t = stream_id; t < N_TILES; t += 128) {
        const int n_base = t * 16;
        v8f acc0 = (v8f){}, acc1 = (v8f){};
        const __bf16* brow = Ebf + (size_t)(n_base + m) * EMBED_DIM;
#pragma unroll
        for (int f = 0; f < 4; ++f) {
            v16bf b = *(const v16bf*)(brow + f * 32 + half * 16);   // shared B
            acc0 = __builtin_amdgcn_wmma_f32_16x16x32_bf16(
                false, afr[0][f], false, b, (short)0, acc0, false, false);
            acc1 = __builtin_amdgcn_wmma_f32_16x16x32_bf16(
                false, afr[1][f], false, b, (short)0, acc1, false, false);
        }
        // Stage both 16x16 C fragments: row-in-block = s*16 + r + 8*half.
#pragma unroll
        for (int r = 0; r < 8; ++r) {
            mystage[(r + 8 * half) * 16 + m]      = acc0[r];
            mystage[(16 + r + 8 * half) * 16 + m] = acc1[r];
        }
        __builtin_amdgcn_wave_barrier();   // DS ops from one wave are in-order

        // All 32 lanes active: lane l scans row l of the 32-row block.
        {
            const float* rowp = mystage + lane * 16;
            for (int c = 0; c < 16; ++c) {
                const float s = rowp[c];
                if (s > curMin) {          // rare after warm-up: ~O(50 ln N)
                    myls[minPos] = s;
                    myli[minPos] = n_base + c;
                    float mn = myls[0]; int mp = 0;
                    for (int q = 1; q < TOP_K; ++q) {
                        const float v = myls[q];
                        if (v < mn) { mn = v; mp = q; }
                    }
                    curMin = mn; minPos = mp;
                }
            }
        }
        __builtin_amdgcn_wave_barrier();
    }

    __syncthreads();
    // Block merge: wave w handles rows w, w+8, w+16, w+24; 400 contiguous
    // candidates per row; 50 rounds of wave-parallel argmax each.
    for (int rr = wave; rr < 32; rr += 8) {
        float* rs = lscore + rr * 8 * TOP_K;   // 400 entries
        int*   ri = lidx   + rr * 8 * TOP_K;
        const size_t obase = ((size_t)(m_base + rr) * N_CHUNKS + chunk) * TOP_K;
        for (int k = 0; k < TOP_K; ++k) {
            float v = -__builtin_inff(); int p = -1;
            for (int e = lane; e < 8 * TOP_K; e += 32) {
                const float s = rs[e];
                if (s > v) { v = s; p = e; }
            }
            wave_argmax(v, p);
            if (lane == 0) { candS[obase + k] = v; candI[obase + k] = ri[p]; }
            if (p >= 0 && (p & 31) == lane) rs[p] = -__builtin_inff();
            __builtin_amdgcn_wave_barrier();
        }
    }
}

// ---------------------------------------------------------------------------
// Final merge: one wave per batch row merges 16 chunks x 50 = 800 candidates
// (staged in LDS) into the final descending top-50 index list.
// ---------------------------------------------------------------------------
__global__ __launch_bounds__(256)
void rec_topk_final(const float* __restrict__ candS,
                    const int* __restrict__ candI,
                    int* __restrict__ out) {
    extern __shared__ float fsm[];
    const int tid  = threadIdx.x;
    const int wave = tid >> 5;
    const int lane = tid & 31;
    const int row  = blockIdx.x * 8 + wave;
    const int NC   = N_CHUNKS * TOP_K;       // 800

    float* ls = fsm + wave * NC;                       // [800] f32
    int*   li = (int*)(fsm + 8 * NC) + wave * NC;      // [800] i32

    const size_t base = (size_t)row * NC;
    for (int e = lane; e < NC; e += 32) {
        ls[e] = candS[base + e];
        li[e] = candI[base + e];
    }
    __builtin_amdgcn_wave_barrier();

    int* orow = out + (size_t)row * TOP_K;
    for (int k = 0; k < TOP_K; ++k) {
        float v = -__builtin_inff(); int p = -1;
        for (int e = lane; e < NC; e += 32) {   // 25 entries per lane
            const float s = ls[e];
            if (s > v) { v = s; p = e; }
        }
        wave_argmax(v, p);
        if (lane == 0) orow[k] = li[p];
        if (p >= 0 && (p & 31) == lane) ls[p] = -__builtin_inff();
        __builtin_amdgcn_wave_barrier();
    }
}

// ---------------------------------------------------------------------------
extern "C" void kernel_launch(void* const* d_in, const int* in_sizes, int n_in,
                              void* d_out, int out_size, void* d_ws, size_t ws_size,
                              hipStream_t stream) {
    const float* x = (const float*)d_in[0];            // (1024, 100000) f32
    const float* E = (const float*)d_in[1];            // (100000, 128) f32
    int* out = (int*)d_out;                            // (1024, 50) int32 indices

    char* ws = (char*)d_ws;
    __bf16* Ebf    = (__bf16*)(ws + WS_EBF_OFF);
    __bf16* ETbf   = (__bf16*)(ws + WS_ETBF_OFF);
    float*  xemb   = (float*)(ws + WS_XEMB_OFF);
    __bf16* xembbf = (__bf16*)(ws + WS_XEMBBF_OFF);
    float*  candS  = (float*)(ws + WS_CANDS_OFF);
    int*    candI  = (int*)(ws + WS_CANDI_OFF);

    rec_zero_xemb<<<(BATCH * EMBED_DIM + 255) / 256, 256, 0, stream>>>(xemb);
    rec_convert_E<<<4096, 256, 0, stream>>>(E, Ebf, ETbf);
    rec_gemm1<<<dim3(32, 125), 32, 0, stream>>>(x, ETbf, xemb);
    rec_xemb_to_bf16<<<(BATCH * EMBED_DIM + 255) / 256, 256, 0, stream>>>(xemb, xembbf);

    const size_t smem_g2 = (8 * 512 + 32 * 8 * TOP_K) * sizeof(float)
                         + 32 * 8 * TOP_K * sizeof(int);          // 118784 B
    rec_gemm2_topk<<<dim3(32, N_CHUNKS), 256, smem_g2, stream>>>(xembbf, Ebf, candS, candI);

    const size_t smem_fm = 8 * (N_CHUNKS * TOP_K) * (sizeof(float) + sizeof(int)); // 51200 B
    rec_topk_final<<<BATCH / 8, 256, smem_fm, stream>>>(candS, candI, out);
}
